// Attnmamba_Crossattnmamba_Model_19404662243342
// MI455X (gfx1250) — compile-verified, matrix operand order checked by reading, not run on hardware
//
#include <hip/hip_runtime.h>
#include <hip/hip_bf16.h>
#include <math.h>

// ---------------- model constants ----------------
#define B_SZ      16
#define T_IN      2048
#define F_IN      80
#define TGTLEN    128
#define VOCAB     5000
#define DUR       5
#define D_MODEL   512
#define D_INNER   1024
#define NHD       16          // NH heads
#define HEADDIM   64
#define D_STATE   64
#define D_CONVW   4
#define CONV_DIM  1152        // D_INNER + 2*D_STATE
#define D_IN_PROJ 2192        // 2*D_INNER + 2*D_STATE + NH
#define D_FF      2048
#define L_ENC     512         // T_IN/4
#define L_CAT     640         // L_ENC + TGTLEN
#define TOK_ENC   (B_SZ*L_ENC)    // 8192
#define TOK_CAT   (B_SZ*L_CAT)    // 10240
#define TOK_TGT   (B_SZ*TGTLEN)   // 2048
#define QCHUNK    64

#define CDIV(a,b) (((a)+(b)-1)/(b))

typedef __attribute__((ext_vector_type(16))) __bf16 v16bf;
typedef __attribute__((ext_vector_type(8)))  __bf16 v8bf;
typedef __attribute__((ext_vector_type(4)))  __bf16 v4bf;
typedef __attribute__((ext_vector_type(8)))  float  v8f;

static __device__ __forceinline__ __bf16 f2bf(float f) {
  unsigned u = __builtin_bit_cast(unsigned, f);
  unsigned r = u + 0x7FFFu + ((u >> 16) & 1u);   // round-to-nearest-even
  unsigned short h = (unsigned short)(r >> 16);
  return __builtin_bit_cast(__bf16, h);
}

static __device__ __forceinline__ float siluf(float x) {
  return x / (1.0f + expf(-x));
}

// ---------------- WMMA GEMM: C = act(A@B + bias)*scale  (opt. +=) ----------------
// A: M x K row-major (M % 32 == 0, K % 32 == 0), B: K x N row-major, C: M x N.
// act: 0 none, 1 relu, 2 exact gelu.
// Block: 128 threads / 4 waves -> 32x64 C tile. A and B(transposed) staged in LDS
// as bf16; each lane's WMMA fragment = two contiguous 16B runs (ds_load_b128).
__global__ void gemm_wmma_kernel(const float* __restrict__ A, const float* __restrict__ Bm,
                                 const float* __restrict__ bias, float* __restrict__ C,
                                 int M, int N, int K, int act, float scale, int accum) {
  __shared__ __bf16 aT[32 * 32];   // [m][k]
  __shared__ __bf16 bT[64 * 32];   // [n][k]  (transposed tile)

  int tid  = threadIdx.x;
  int lane = tid & 31;
  int wv   = tid >> 5;
  long row0 = (long)blockIdx.y * 32;
  long col0 = (long)blockIdx.x * 64;
  int mr = lane & 15;
  int kh = lane >> 4;
  int wn = wv * 16;

  // A loader: 8 consecutive floats per thread (two float4)
  int rowA = tid >> 2;             // 0..31
  int kA   = (tid & 3) * 8;        // 0,8,16,24
  // B loader: 4x4 block (4 k-rows x 4 n-cols) per thread
  int kB = (tid >> 4) * 4;         // 0..28
  int nB = (tid & 15) * 4;         // 0..60

  bool fastB = ((col0 + 64) <= (long)N) && ((N & 3) == 0);

  v8f acc0, acc1;
#pragma unroll
  for (int r = 0; r < 8; r++) { acc0[r] = 0.0f; acc1[r] = 0.0f; }

  for (int k0 = 0; k0 < K; k0 += 32) {
    // ---- stage A (32x32 f32 -> bf16) ----
    const float* Ap = A + (row0 + rowA) * K + k0 + kA;
    float4 a0 = *(const float4*)Ap;
    float4 a1 = *(const float4*)(Ap + 4);
    v4bf pa0 = { f2bf(a0.x), f2bf(a0.y), f2bf(a0.z), f2bf(a0.w) };
    v4bf pa1 = { f2bf(a1.x), f2bf(a1.y), f2bf(a1.z), f2bf(a1.w) };
    *(v4bf*)&aT[rowA * 32 + kA]     = pa0;
    *(v4bf*)&aT[rowA * 32 + kA + 4] = pa1;
    if (k0 + 32 < K) __builtin_prefetch(Ap + 32, 0, 1);

    // ---- stage B (32x64 f32 -> bf16, transposed to [n][k]) ----
    float rr[4][4];
    if (fastB) {
      const float* Bp = Bm + (long)(k0 + kB) * N + col0 + nB;
#pragma unroll
      for (int i = 0; i < 4; i++) {
        float4 t = *(const float4*)(Bp + (long)i * N);
        rr[i][0] = t.x; rr[i][1] = t.y; rr[i][2] = t.z; rr[i][3] = t.w;
      }
      if (k0 + 32 < K) __builtin_prefetch(Bp + 32LL * N, 0, 1);
    } else {
#pragma unroll
      for (int i = 0; i < 4; i++)
        for (int j = 0; j < 4; j++) {
          long n = col0 + nB + j;
          rr[i][j] = (n < N) ? Bm[(long)(k0 + kB + i) * N + n] : 0.0f;
        }
    }
#pragma unroll
    for (int j = 0; j < 4; j++) {
      v4bf pb = { f2bf(rr[0][j]), f2bf(rr[1][j]), f2bf(rr[2][j]), f2bf(rr[3][j]) };
      *(v4bf*)&bT[(nB + j) * 32 + kB] = pb;
    }
    __syncthreads();

    // ---- fragments: two contiguous 16B K-runs each ----
    v8bf al0 = *(v8bf*)&aT[(mr)      * 32 + kh * 8];
    v8bf ah0 = *(v8bf*)&aT[(mr)      * 32 + 16 + kh * 8];
    v8bf al1 = *(v8bf*)&aT[(mr + 16) * 32 + kh * 8];
    v8bf ah1 = *(v8bf*)&aT[(mr + 16) * 32 + 16 + kh * 8];
    v8bf bl  = *(v8bf*)&bT[(wn + mr) * 32 + kh * 8];
    v8bf bh  = *(v8bf*)&bT[(wn + mr) * 32 + 16 + kh * 8];
    v16bf af0 = __builtin_shufflevector(al0, ah0, 0,1,2,3,4,5,6,7,8,9,10,11,12,13,14,15);
    v16bf af1 = __builtin_shufflevector(al1, ah1, 0,1,2,3,4,5,6,7,8,9,10,11,12,13,14,15);
    v16bf bf  = __builtin_shufflevector(bl,  bh,  0,1,2,3,4,5,6,7,8,9,10,11,12,13,14,15);

    acc0 = __builtin_amdgcn_wmma_f32_16x16x32_bf16(false, af0, false, bf, (short)0, acc0,
                                                   false, false);
    acc1 = __builtin_amdgcn_wmma_f32_16x16x32_bf16(false, af1, false, bf, (short)0, acc1,
                                                   false, false);
    __syncthreads();
  }

  long n = col0 + wn + mr;
  if (n < N) {
    float bv = bias[n];
#pragma unroll
    for (int r = 0; r < 8; r++) {
      long m0 = row0 + r + kh * 8;
      long m1 = m0 + 16;
      float v0 = acc0[r] + bv;
      float v1 = acc1[r] + bv;
      if (act == 1)      { v0 = fmaxf(v0, 0.0f); v1 = fmaxf(v1, 0.0f); }
      else if (act == 2) { v0 = 0.5f * v0 * (1.0f + erff(v0 * 0.70710678118654752f));
                           v1 = 0.5f * v1 * (1.0f + erff(v1 * 0.70710678118654752f)); }
      v0 *= scale; v1 *= scale;
      if (accum) { C[m0 * N + n] += v0; C[m1 * N + n] += v1; }
      else       { C[m0 * N + n]  = v0; C[m1 * N + n]  = v1; }
    }
  }
}

// ---------------- conv frontend ----------------
__global__ void conv3x3_s2_relu(const float* __restrict__ in, const float* __restrict__ w,
                                const float* __restrict__ bias, float* __restrict__ out,
                                int B, int Cin, int H, int W, int Cout) {
  int Ho = H / 2, Wo = W / 2;
  long idx = (long)blockIdx.x * blockDim.x + threadIdx.x;
  long total = (long)B * Cout * Ho * Wo;
  if (idx >= total) return;
  int x = idx % Wo; long r = idx / Wo;
  int y = r % Ho;  r /= Ho;
  int co = r % Cout; int b = r / Cout;
  float acc = bias[co];
  for (int ci = 0; ci < Cin; ci++)
    for (int ky = 0; ky < 3; ky++) {
      int iy = 2 * y - 1 + ky; if (iy < 0 || iy >= H) continue;
      for (int kx = 0; kx < 3; kx++) {
        int ix = 2 * x - 1 + kx; if (ix < 0 || ix >= W) continue;
        acc += in[((long)(b * Cin + ci) * H + iy) * W + ix] *
               w[((co * Cin + ci) * 3 + ky) * 3 + kx];
      }
    }
  out[idx] = fmaxf(acc, 0.0f);
}

// [16][32][512][20] -> [8192][640], feat = c*20+f
__global__ void transpose_feat(const float* __restrict__ in, float* __restrict__ out) {
  long idx = (long)blockIdx.x * blockDim.x + threadIdx.x;
  long total = (long)TOK_ENC * 640;
  if (idx >= total) return;
  int feat = idx % 640; long tok = idx / 640;
  int c = feat / 20, f = feat % 20;
  int b = tok / L_ENC, t = tok % L_ENC;
  out[idx] = in[((long)(b * 32 + c) * L_ENC + t) * 20 + f];
}

// ---------------- masks ----------------
__global__ void compute_masks(const int* __restrict__ il, const int* __restrict__ tl,
                              float* __restrict__ mask, float* __restrict__ tm,
                              float* __restrict__ fm) {
  int tid = threadIdx.x;
  for (int e = tid; e < B_SZ * L_ENC; e += blockDim.x) {
    int b = e / L_ENC, t = e % L_ENC;
    int ol = il[b]; ol = (ol - 1) / 2 + 1; ol = (ol - 1) / 2 + 1;
    mask[e] = (t >= L_ENC - ol) ? 1.0f : 0.0f;
  }
  for (int e = tid; e < B_SZ * TGTLEN; e += blockDim.x) {
    int b = e / TGTLEN, t = e % TGTLEN;
    tm[e] = (t >= TGTLEN - tl[b]) ? 1.0f : 0.0f;
  }
  __syncthreads();
  for (int e = tid; e < B_SZ * L_CAT; e += blockDim.x) {
    int b = e / L_CAT, t = e % L_CAT;
    fm[e] = (t < L_ENC) ? mask[b * L_ENC + t] : tm[b * TGTLEN + t - L_ENC];
  }
}

// ---------------- elementwise / norms ----------------
__global__ void scale_mask(float* __restrict__ x, const float* __restrict__ mask,
                           long total, int dim) {
  long idx = (long)blockIdx.x * blockDim.x + threadIdx.x;
  if (idx >= total) return;
  x[idx] *= mask[idx / dim];
}

__global__ void rms_norm_k(const float* __restrict__ in, float* __restrict__ out, int dim) {
  __shared__ float red[256];
  long t = blockIdx.x;
  const float* x = in + t * dim;
  float* y = out + t * dim;
  float ss = 0.0f;
  for (int d = threadIdx.x; d < dim; d += 256) { float v = x[d]; ss += v * v; }
  red[threadIdx.x] = ss; __syncthreads();
  for (int s = 128; s > 0; s >>= 1) { if (threadIdx.x < s) red[threadIdx.x] += red[threadIdx.x + s]; __syncthreads(); }
  float rs = rsqrtf(red[0] / (float)dim + 1e-6f);
  for (int d = threadIdx.x; d < dim; d += 256) y[d] = x[d] * rs;
}

__global__ void layer_norm_k(const float* __restrict__ in, const float* __restrict__ w,
                             const float* __restrict__ b, float* __restrict__ out, int dim) {
  __shared__ float red[256];
  __shared__ float smean, svar;
  long t = blockIdx.x;
  const float* x = in + t * dim;
  float* y = out + t * dim;
  float s = 0.0f;
  for (int d = threadIdx.x; d < dim; d += 256) s += x[d];
  red[threadIdx.x] = s; __syncthreads();
  for (int k = 128; k > 0; k >>= 1) { if (threadIdx.x < k) red[threadIdx.x] += red[threadIdx.x + k]; __syncthreads(); }
  if (threadIdx.x == 0) smean = red[0] / (float)dim;
  __syncthreads();
  float m = smean, v = 0.0f;
  for (int d = threadIdx.x; d < dim; d += 256) { float e = x[d] - m; v += e * e; }
  red[threadIdx.x] = v; __syncthreads();
  for (int k = 128; k > 0; k >>= 1) { if (threadIdx.x < k) red[threadIdx.x] += red[threadIdx.x + k]; __syncthreads(); }
  if (threadIdx.x == 0) svar = red[0] / (float)dim;
  __syncthreads();
  float rs = rsqrtf(svar + 1e-5f);
  for (int d = threadIdx.x; d < dim; d += 256) y[d] = (x[d] - m) * rs * w[d] + b[d];
}

// ---------------- mamba pieces ----------------
// causal depthwise conv (width 4) over xBC slice of zx, then silu
__global__ void dwconv_silu(const float* __restrict__ zx, const float* __restrict__ w,
                            const float* __restrict__ bias, float* __restrict__ out,
                            int B, int L) {
  long idx = (long)blockIdx.x * blockDim.x + threadIdx.x;
  long total = (long)B * L * CONV_DIM;
  if (idx >= total) return;
  int c = idx % CONV_DIM; long r = idx / CONV_DIM;
  int t = r % L; int b = r / L;
  float acc = bias[c];
  for (int k = 0; k < D_CONVW; k++) {
    int ts = t - (D_CONVW - 1) + k;
    if (ts >= 0)
      acc += zx[(long)(b * L + ts) * D_IN_PROJ + D_INNER + c] * w[k * CONV_DIM + c];
  }
  out[idx] = siluf(acc);
}

__global__ void dt_softplus(const float* __restrict__ zx, const float* __restrict__ dtb,
                            const float* __restrict__ mask, float* __restrict__ dt, int ntok) {
  long idx = (long)blockIdx.x * blockDim.x + threadIdx.x;
  if (idx >= (long)ntok * NHD) return;
  int h = idx % NHD; long t = idx / NHD;
  float x = zx[t * D_IN_PROJ + (D_IN_PROJ - NHD) + h] + dtb[h];
  float sp = (x > 20.0f) ? x : log1pf(expf(x));
  dt[idx] = sp * mask[t];
}

// cum[b][c][i][h] = cumsum_i( dt * -exp(A_log) ), with optional time reversal
__global__ void ssd_cumsum(const float* __restrict__ dt, const float* __restrict__ A_log,
                           float* __restrict__ cum, int B, int L, int rev) {
  int nc = L / QCHUNK;
  long idx = (long)blockIdx.x * blockDim.x + threadIdx.x;
  if (idx >= (long)B * nc * NHD) return;
  int h = idx % NHD; int c = (idx / NHD) % nc; int b = idx / (NHD * nc);
  float A = -expf(A_log[h]);
  float run = 0.0f;
  for (int i = 0; i < QCHUNK; i++) {
    int t = c * QCHUNK + i; int tt = rev ? (L - 1 - t) : t;
    run += dt[(long)(b * L + tt) * NHD + h] * A;
    cum[(((long)(b * nc + c) * QCHUNK) + i) * NHD + h] = run;
  }
}

// intra-chunk Yd + D*x.  grid: B*nc*NH blocks of 64 threads (thread = row i)
__global__ void ssd_intra(const float* __restrict__ xBC, const float* __restrict__ dt,
                          const float* __restrict__ cum, const float* __restrict__ Dp,
                          float* __restrict__ Y, int B, int L, int rev, int accum) {
  __shared__ float sB[QCHUNK][QCHUNK];   // B rows [j][n]
  __shared__ float sX[QCHUNK][QCHUNK];   // xdt   [j][p]
  __shared__ float sW[QCHUNK][QCHUNK];   // CB*decay [i][j]
  __shared__ float scum[QCHUNK];
  int nc = L / QCHUNK;
  int h = blockIdx.x % NHD;
  int c = (blockIdx.x / NHD) % nc;
  int b = blockIdx.x / (NHD * nc);
  int i = threadIdx.x;
  int t_i = c * QCHUNK + i;
  int tt_i = rev ? (L - 1 - t_i) : t_i;
  long rowi = (long)(b * L + tt_i) * CONV_DIM;
  for (int n = 0; n < D_STATE; n++) sB[i][n] = xBC[rowi + D_INNER + n];
  float dti = dt[(long)(b * L + tt_i) * NHD + h];
  for (int p = 0; p < HEADDIM; p++) sX[i][p] = xBC[rowi + h * HEADDIM + p] * dti;
  scum[i] = cum[(((long)(b * nc + c) * QCHUNK) + i) * NHD + h];
  float cRow[D_STATE];
  for (int n = 0; n < D_STATE; n++) cRow[n] = xBC[rowi + D_INNER + D_STATE + n];
  __syncthreads();
  float ci = scum[i];
  for (int j = 0; j <= i; j++) {
    float cb = 0.0f;
    for (int n = 0; n < D_STATE; n++) cb += cRow[n] * sB[j][n];
    sW[i][j] = cb * expf(ci - scum[j]);
  }
  __syncthreads();
  float Dh = Dp[h];
  for (int p = 0; p < HEADDIM; p++) {
    float acc = 0.0f;
    for (int j = 0; j <= i; j++) acc += sW[i][j] * sX[j][p];
    float out = acc + Dh * xBC[rowi + h * HEADDIM + p];
    long yo = ((long)(b * L + tt_i) * NHD + h) * HEADDIM + p;
    if (accum) Y[yo] += out; else Y[yo] = out;
  }
}

// inter-chunk scan + Yo.  grid: B*NH blocks of 256 threads, chunks sequential, state in LDS
__global__ void ssd_scan(const float* __restrict__ xBC, const float* __restrict__ dt,
                         const float* __restrict__ cum, float* __restrict__ Y,
                         int B, int L, int rev) {
  __shared__ float sH[HEADDIM][D_STATE];   // state [p][n]
  __shared__ float sCB[QCHUNK][D_STATE];   // C rows, then B rows
  __shared__ float sX[QCHUNK][HEADDIM];    // xdt [j][p]
  __shared__ float scum[QCHUNK];
  __shared__ float sdend[QCHUNK];
  int h = blockIdx.x % NHD;
  int b = blockIdx.x / NHD;
  int tid = threadIdx.x;
  int nc = L / QCHUNK;
  for (int e = tid; e < HEADDIM * D_STATE; e += 256) sH[e >> 6][e & 63] = 0.0f;
  __syncthreads();
  for (int c = 0; c < nc; c++) {
    if (tid < QCHUNK) scum[tid] = cum[(((long)(b * nc + c) * QCHUNK) + tid) * NHD + h];
    for (int e = tid; e < QCHUNK * D_STATE; e += 256) {
      int i = e >> 6, n = e & 63;
      int t = c * QCHUNK + i; int tt = rev ? (L - 1 - t) : t;
      sCB[i][n] = xBC[(long)(b * L + tt) * CONV_DIM + D_INNER + D_STATE + n];
    }
    __syncthreads();
    // Yo from previous-chunk state
    for (int e = tid; e < QCHUNK * HEADDIM; e += 256) {
      int i = e >> 6, p = e & 63;
      float a = 0.0f;
      for (int n = 0; n < D_STATE; n++) a += sCB[i][n] * sH[p][n];
      int t = c * QCHUNK + i; int tt = rev ? (L - 1 - t) : t;
      Y[((long)(b * L + tt) * NHD + h) * HEADDIM + p] += expf(scum[i]) * a;
    }
    __syncthreads();
    float last = scum[QCHUNK - 1];
    if (tid < QCHUNK) sdend[tid] = expf(last - scum[tid]);
    for (int e = tid; e < QCHUNK * D_STATE; e += 256) {
      int j = e >> 6, n = e & 63;
      int t = c * QCHUNK + j; int tt = rev ? (L - 1 - t) : t;
      sCB[j][n] = xBC[(long)(b * L + tt) * CONV_DIM + D_INNER + n];
    }
    for (int e = tid; e < QCHUNK * HEADDIM; e += 256) {
      int j = e >> 6, p = e & 63;
      int t = c * QCHUNK + j; int tt = rev ? (L - 1 - t) : t;
      sX[j][p] = xBC[(long)(b * L + tt) * CONV_DIM + h * HEADDIM + p] *
                 dt[(long)(b * L + tt) * NHD + h];
    }
    __syncthreads();
    float cdec = expf(last);
    for (int e = tid; e < HEADDIM * D_STATE; e += 256) {
      int p = e >> 6, n = e & 63;
      float s = 0.0f;
      for (int j = 0; j < QCHUNK; j++) s += sCB[j][n] * sdend[j] * sX[j][p];
      sH[p][n] = sH[p][n] * cdec + s;
    }
    __syncthreads();
  }
}

// g = rms(y * silu(z)) * norm_w  (in-place on y), block per token
__global__ void gate_rms(float* __restrict__ y, const float* __restrict__ zx,
                         const float* __restrict__ nw) {
  __shared__ float red[256];
  long t = blockIdx.x;
  const float* z = zx + t * D_IN_PROJ;
  float* yr = y + t * D_INNER;
  float ss = 0.0f;
  for (int d = threadIdx.x; d < D_INNER; d += 256) {
    float v = yr[d] * siluf(z[d]); ss += v * v;
  }
  red[threadIdx.x] = ss; __syncthreads();
  for (int s = 128; s > 0; s >>= 1) { if (threadIdx.x < s) red[threadIdx.x] += red[threadIdx.x + s]; __syncthreads(); }
  float rs = rsqrtf(red[0] / (float)D_INNER + 1e-6f);
  for (int d = threadIdx.x; d < D_INNER; d += 256) {
    float v = yr[d] * siluf(z[d]);
    yr[d] = v * rs * nw[d];
  }
}

// ---------------- decoder glue ----------------
__global__ void embed_k(const int* __restrict__ tg, const float* __restrict__ emb,
                        float* __restrict__ out) {
  long idx = (long)blockIdx.x * blockDim.x + threadIdx.x;
  long total = (long)TOK_TGT * D_MODEL;
  if (idx >= total) return;
  int d = idx % D_MODEL; long t = idx / D_MODEL;
  out[idx] = emb[(long)tg[t] * D_MODEL + d];
}

__global__ void build_cat(const float* __restrict__ enc, const float* __restrict__ t1,
                          const float* __restrict__ mask, float* __restrict__ cat) {
  long idx = (long)blockIdx.x * blockDim.x + threadIdx.x;
  long total = (long)TOK_CAT * D_MODEL;
  if (idx >= total) return;
  int d = idx % D_MODEL; long r = idx / D_MODEL;
  int t = r % L_CAT; int b = r / L_CAT;
  if (t < L_ENC)
    cat[idx] = enc[((long)(b * L_ENC + t)) * D_MODEL + d] * mask[b * L_ENC + t];
  else
    cat[idx] = t1[((long)(b * TGTLEN + (t - L_ENC))) * D_MODEL + d];
}

__global__ void add_slice(float* __restrict__ tgt, const float* __restrict__ hbuf) {
  long idx = (long)blockIdx.x * blockDim.x + threadIdx.x;
  long total = (long)TOK_TGT * D_MODEL;
  if (idx >= total) return;
  int d = idx % D_MODEL; long r = idx / D_MODEL;
  int t = r % TGTLEN; int b = r / TGTLEN;
  tgt[idx] += hbuf[((long)(b * L_CAT + L_ENC + t)) * D_MODEL + d];
}

// ---------------- host orchestration ----------------
#define PF(i) ((const float*)d_in[(i)])
#define PI(i) ((const int*)d_in[(i)])

// param leaf indices (dict insertion order)
enum {
  IN_X = 0,
  P_CONV1_W = 1, P_CONV1_B, P_CONV2_W, P_CONV2_B, P_PROJ_W, P_PROJ_B,
  P_ENC0 = 7,                    // 3 layers x 14 leaves
  P_ENC_LN_W = 49, P_ENC_LN_B = 50, P_CTC_W = 51, P_CTC_B = 52, P_EMB = 53,
  P_DEC0 = 54,
  P_DEC_LN_W = 96, P_DEC_LN_B = 97, P_AED_W = 98, P_AED_B = 99,
  IN_ILEN = 100, IN_TGT = 101, IN_TLEN = 102
};
// within-layer offsets
enum { M_IN_W = 0, M_IN_B, M_CONV_W, M_CONV_B, M_DT_BIAS, M_A_LOG, M_D, M_NORM_W,
       M_OUT_W, M_OUT_B, L_W1, L_B1, L_W2, L_B2 };

static void run_gemm(hipStream_t s, const float* A, const float* B, const float* bias,
                     float* C, int M, int N, int K, int act, float scale, int accum) {
  dim3 blk(128);
  dim3 grd(CDIV(N, 64), M / 32);
  gemm_wmma_kernel<<<grd, blk, 0, s>>>(A, B, bias, C, M, N, K, act, scale, accum);
}

static void run_mamba(hipStream_t s, void* const* d_in, int base,
                      const float* inp, const float* mask,
                      int B, int L, int bidir,
                      float* dst, float out_scale, int out_accum,
                      float* bufZX, float* bufConv, float* bufY, float* bufDT,
                      float* bufCum) {
  int tok = B * L;
  int nc = L / QCHUNK;
  run_gemm(s, inp, PF(base + M_IN_W), PF(base + M_IN_B), bufZX, tok, D_IN_PROJ, D_MODEL,
           0, 1.0f, 0);
  dwconv_silu<<<CDIV((long)tok * CONV_DIM, 256), 256, 0, s>>>(
      bufZX, PF(base + M_CONV_W), PF(base + M_CONV_B), bufConv, B, L);
  dt_softplus<<<CDIV(tok * NHD, 256), 256, 0, s>>>(bufZX, PF(base + M_DT_BIAS), mask,
                                                   bufDT, tok);
  for (int dir = 0; dir < (bidir ? 2 : 1); dir++) {
    ssd_cumsum<<<CDIV(B * nc * NHD, 128), 128, 0, s>>>(bufDT, PF(base + M_A_LOG), bufCum,
                                                       B, L, dir);
    ssd_intra<<<B * nc * NHD, QCHUNK, 0, s>>>(bufConv, bufDT, bufCum, PF(base + M_D),
                                              bufY, B, L, dir, dir);
    ssd_scan<<<B * NHD, 256, 0, s>>>(bufConv, bufDT, bufCum, bufY, B, L, dir);
  }
  gate_rms<<<tok, 256, 0, s>>>(bufY, bufZX, PF(base + M_NORM_W));
  run_gemm(s, bufY, PF(base + M_OUT_W), PF(base + M_OUT_B), dst, tok, D_MODEL, D_INNER,
           0, out_scale, out_accum);
}

extern "C" void kernel_launch(void* const* d_in, const int* in_sizes, int n_in,
                              void* d_out, int out_size, void* d_ws, size_t ws_size,
                              hipStream_t stream) {
  (void)in_sizes; (void)n_in; (void)out_size; (void)ws_size;
  float* ws = (float*)d_ws;
  size_t o = 0;
  auto alloc = [&](size_t n) { float* p = ws + o; o += (n + 63) & ~(size_t)63; return p; };

  float* bufU    = alloc((size_t)TOK_CAT * D_MODEL);     // rms'd inputs / h
  float* bufZX   = alloc((size_t)TOK_CAT * D_IN_PROJ);   // also conv1-out / mlp-hidden
  float* bufConv = alloc((size_t)TOK_CAT * CONV_DIM);    // also conv2-out
  float* bufY    = alloc((size_t)TOK_CAT * D_INNER);     // also transposed frontend
  float* bufEnc  = alloc((size_t)TOK_ENC * D_MODEL);
  float* bufTgt  = alloc((size_t)TOK_TGT * D_MODEL);
  float* bufCat  = alloc((size_t)TOK_CAT * D_MODEL);
  float* bufDT   = alloc((size_t)TOK_CAT * NHD);
  float* bufCum  = alloc((size_t)B_SZ * (L_CAT / QCHUNK) * QCHUNK * NHD);
  float* mMask   = alloc(B_SZ * L_ENC);
  float* mTm     = alloc(B_SZ * TGTLEN);
  float* mFm     = alloc(B_SZ * L_CAT);

  float* outCtc = (float*)d_out;
  float* outAed = outCtc + (size_t)TOK_ENC * VOCAB;

  // ---- frontend ----
  conv3x3_s2_relu<<<CDIV((long)B_SZ * 32 * (T_IN/2) * (F_IN/2), 256), 256, 0, stream>>>(
      PF(IN_X), PF(P_CONV1_W), PF(P_CONV1_B), bufZX, B_SZ, 1, T_IN, F_IN, 32);
  conv3x3_s2_relu<<<CDIV((long)B_SZ * 32 * L_ENC * 20, 256), 256, 0, stream>>>(
      bufZX, PF(P_CONV2_W), PF(P_CONV2_B), bufConv, B_SZ, 32, T_IN/2, F_IN/2, 32);
  transpose_feat<<<CDIV((long)TOK_ENC * 640, 256), 256, 0, stream>>>(bufConv, bufY);
  run_gemm(stream, bufY, PF(P_PROJ_W), PF(P_PROJ_B), bufEnc, TOK_ENC, D_MODEL, 640,
           0, 1.0f, 0);
  compute_masks<<<1, 256, 0, stream>>>(PI(IN_ILEN), PI(IN_TLEN), mMask, mTm, mFm);

  // ---- encoder ----
  for (int l = 0; l < 3; l++) {
    int base = P_ENC0 + 14 * l;
    scale_mask<<<CDIV((long)TOK_ENC * D_MODEL, 256), 256, 0, stream>>>(
        bufEnc, mMask, (long)TOK_ENC * D_MODEL, D_MODEL);
    rms_norm_k<<<TOK_ENC, 256, 0, stream>>>(bufEnc, bufU, D_MODEL);
    run_mamba(stream, d_in, base, bufU, mMask, B_SZ, L_ENC, /*bidir=*/1,
              bufEnc, 2.0f, 0, bufZX, bufConv, bufY, bufDT, bufCum);
    rms_norm_k<<<TOK_ENC, 256, 0, stream>>>(bufEnc, bufU, D_MODEL);
    run_gemm(stream, bufU, PF(base + L_W1), PF(base + L_B1), bufZX, TOK_ENC, D_FF,
             D_MODEL, /*gelu*/2, 1.0f, 0);
    run_gemm(stream, bufZX, PF(base + L_W2), PF(base + L_B2), bufEnc, TOK_ENC, D_MODEL,
             D_FF, 0, 1.0f, /*accum*/1);
  }
  layer_norm_k<<<TOK_ENC, 256, 0, stream>>>(bufEnc, PF(P_ENC_LN_W), PF(P_ENC_LN_B),
                                            bufEnc, D_MODEL);
  run_gemm(stream, bufEnc, PF(P_CTC_W), PF(P_CTC_B), outCtc, TOK_ENC, VOCAB, D_MODEL,
           0, 1.0f, 0);

  // ---- decoder ----
  embed_k<<<CDIV((long)TOK_TGT * D_MODEL, 256), 256, 0, stream>>>(PI(IN_TGT), PF(P_EMB),
                                                                  bufTgt);
  for (int l = 0; l < 3; l++) {
    int base = P_DEC0 + 14 * l;
    scale_mask<<<CDIV((long)TOK_TGT * D_MODEL, 256), 256, 0, stream>>>(
        bufTgt, mTm, (long)TOK_TGT * D_MODEL, D_MODEL);
    rms_norm_k<<<TOK_TGT, 256, 0, stream>>>(bufTgt, bufU, D_MODEL);
    build_cat<<<CDIV((long)TOK_CAT * D_MODEL, 256), 256, 0, stream>>>(bufEnc, bufU, mMask,
                                                                      bufCat);
    run_mamba(stream, d_in, base, bufCat, mFm, B_SZ, L_CAT, /*bidir=*/0,
              bufU, 1.0f, 0, bufZX, bufConv, bufY, bufDT, bufCum);
    add_slice<<<CDIV((long)TOK_TGT * D_MODEL, 256), 256, 0, stream>>>(bufTgt, bufU);
    rms_norm_k<<<TOK_TGT, 256, 0, stream>>>(bufTgt, bufU, D_MODEL);
    run_gemm(stream, bufU, PF(base + L_W1), PF(base + L_B1), bufZX, TOK_TGT, D_FF,
             D_MODEL, 2, 1.0f, 0);
    run_gemm(stream, bufZX, PF(base + L_W2), PF(base + L_B2), bufTgt, TOK_TGT, D_MODEL,
             D_FF, 0, 1.0f, 1);
  }
  layer_norm_k<<<TOK_TGT, 256, 0, stream>>>(bufTgt, PF(P_DEC_LN_W), PF(P_DEC_LN_B),
                                            bufTgt, D_MODEL);
  run_gemm(stream, bufTgt, PF(P_AED_W), PF(P_AED_B), outAed, TOK_TGT, VOCAB + DUR,
           D_MODEL, 0, 1.0f, 0);
}
